// Model_49022756717271
// MI455X (gfx1250) — matrix operand order repaired and checked by reading, not compiled, not run on hardware
//
#include <hip/hip_runtime.h>
#include <hip/hip_bf16.h>

// CDNA5 WMMA fragment types (wave32)
typedef __attribute__((ext_vector_type(16))) __bf16 v16bf;
typedef __attribute__((ext_vector_type(8)))  __bf16 v8bf;
typedef __attribute__((ext_vector_type(8)))  float  v8f;

#define BN_EPS 1e-4f

// ---------------------------------------------------------------------------
// f32 -> bf16 weight conversion
// ---------------------------------------------------------------------------
__global__ void cvt_bf16_kernel(const float* __restrict__ w, __bf16* __restrict__ o, int n) {
  int i = blockIdx.x * 256 + threadIdx.x;
  if (i < n) o[i] = (__bf16)w[i];
}

__global__ void zero_kernel(float* __restrict__ p, int n) {
  int i = blockIdx.x * 128 + threadIdx.x;
  if (i < n) p[i] = 0.f;
}

// ---------------------------------------------------------------------------
// BatchNorm statistics: per-channel sum and sum of squares.
// If 256 % C == 0, each thread's channel is grid-stride-invariant, so partials
// accumulate in VGPRs and touch LDS (ds_add_f32) only once at the end.
// ---------------------------------------------------------------------------
template <int C>
__global__ void bn_stats_kernel(const float* __restrict__ x, float* __restrict__ sums,
                                int N) {
  __shared__ float ssum[64];
  __shared__ float ssq[64];
  int tid = threadIdx.x;
  if (tid < 64) { ssum[tid] = 0.f; ssq[tid] = 0.f; }
  __syncthreads();
  int total  = N * C;
  int stride = gridDim.x * 256;
  int i0     = blockIdx.x * 256 + tid;
  if constexpr (256 % C == 0) {
    // stride % C == 0 -> channel fixed per thread
    int c = i0 % C;
    float s = 0.f, q = 0.f;
    for (int i = i0; i < total; i += stride) {
      float v = x[i];
      s += v;
      q += v * v;
    }
    atomicAdd(&ssum[c], s);
    atomicAdd(&ssq[c], q);
  } else {
    for (int i = i0; i < total; i += stride) {
      float v = x[i];
      int c = i % C;
      atomicAdd(&ssum[c], v);
      atomicAdd(&ssq[c], v * v);
    }
  }
  __syncthreads();
  if (tid < C) {
    atomicAdd(&sums[tid],      ssum[tid]);
    atomicAdd(&sums[64 + tid], ssq[tid]);
  }
}

// scale = g * rsqrt(var + eps) ; bias = b - mean*scale
__global__ void bn_finalize_kernel(const float* __restrict__ sums,
                                   const float* __restrict__ g, const float* __restrict__ b,
                                   float* __restrict__ scale, float* __restrict__ bias,
                                   float invN, int C) {
  int c = threadIdx.x;
  if (c < C) {
    float mean = sums[c] * invN;
    float var  = sums[64 + c] * invN - mean * mean;
    float s    = g[c] * rsqrtf(var + BN_EPS);
    scale[c] = s;
    bias[c]  = b[c] - mean * s;
  }
}

// Elementwise BN+ReLU to f32 (only needed to materialize block0's shortcut)
template <int C>
__global__ void bnrelu_kernel(const float* __restrict__ x,
                              const float* __restrict__ scale, const float* __restrict__ bias,
                              float* __restrict__ y, int total) {
  int i = blockIdx.x * 256 + threadIdx.x;
  if (i < total) {
    int c = i % C;
    float v = fmaf(x[i], scale[c], bias[c]);
    y[i] = v > 0.f ? v : 0.f;
  }
}

// MaxPool 3x3 stride 2 VALID: [512,95,95,8] -> [512,47,47,8] (all constant)
__global__ void maxpool_kernel(const float* __restrict__ in, float* __restrict__ out) {
  constexpr int C = 8, HI = 95, WI = 95, HO = 47, WO = 47;
  int i = blockIdx.x * 256 + threadIdx.x;
  if (i >= 512 * HO * WO * C) return;
  int c  = i % C;
  int t  = i / C;
  int wo = t % WO; t /= WO;
  int ho = t % HO;
  int b  = t / HO;
  float m = -3.402823466e38f;
#pragma unroll
  for (int kh = 0; kh < 3; ++kh)
#pragma unroll
    for (int kw = 0; kw < 3; ++kw) {
      int hi = ho * 2 + kh, wi = wo * 2 + kw;
      float v = in[((b * HI + hi) * WI + wi) * C + c];
      m = v > m ? v : m;
    }
  out[i] = m;
}

// ---------------------------------------------------------------------------
// Implicit-GEMM convolution with WMMA bf16 (f32 accumulate).
//   M = B*HO*WO output sites, N = COUT, K = KH*KW*CIN  (all compile-time dims)
// Fusions: BN(scale,bias)+ReLU on A while staging; residual add and per-column
// bias in epilogue. Block = 256 threads = 8 wave32 waves, each owning a 16-row
// M subtile. A staged row-major, B staged TRANSPOSED ([n][k]) so both WMMA
// fragments are fetched with 2x ds_read_b128 per operand. For even CIN the
// A-stage pair is a single contiguous float2 global load.
// ---------------------------------------------------------------------------
template <int CIN, int COUT, int KH, int KW, int S, int PAD,
          int HI, int WI, int HO, int WO>
__global__ __launch_bounds__(256)
void conv_wmma_kernel(const float* __restrict__ in, const __bf16* __restrict__ wt,
                      const float* __restrict__ bnscale, const float* __restrict__ bnbias,
                      const float* __restrict__ residual, const float* __restrict__ colbias,
                      float* __restrict__ out, int Bn) {
  constexpr int K      = KH * KW * CIN;
  constexpr int KC     = 64;
  constexpr int NCHUNK = (K + KC - 1) / KC;
  constexpr int NT     = (COUT + 15) / 16;
  constexpr int NB     = NT * 16;
  constexpr int LDA    = KC + 8;   // 72 bf16 = 144B rows: 16B aligned, bank-staggered
  constexpr int LDB    = KC + 8;
  constexpr int HOWO   = HO * WO;

  __shared__ __align__(16) __bf16 lA[128 * LDA];
  __shared__ __align__(16) __bf16 lBt[NB * LDB];   // transposed: [n][k]

  const int tid   = threadIdx.x;
  const int wave  = tid >> 5;
  const int lane  = tid & 31;
  const int lhalf = lane >> 4;   // 0 or 1
  const int l16   = lane & 15;

  const int M     = Bn * HOWO;
  const int mBase = (int)blockIdx.x * 128;

  v8f acc[NT];
#pragma unroll
  for (int t = 0; t < NT; ++t)
#pragma unroll
    for (int e = 0; e < 8; ++e) acc[t][e] = 0.f;

  for (int kc = 0; kc < NCHUNK; ++kc) {
    __syncthreads();
    // ---- stage A (im2col, fused BN+ReLU, f32 -> bf16 pairs -> ds_write_b32)
    for (int idx = tid; idx < 128 * (KC / 2); idx += 256) {
      int row = idx / (KC / 2);
      int kp  = (idx % (KC / 2)) * 2;
      int m   = mBase + row;
      bool mOk = m < M;
      int b = 0, ho = 0, wo = 0;
      if (mOk) {
        b = m / HOWO;
        int rem = m % HOWO;
        ho = rem / WO;
        wo = rem % WO;
      }
      union { __bf16 h[2]; unsigned u; } p;
      if constexpr ((CIN & 1) == 0) {
        // even CIN: pair (k,k+1) shares the spatial tap, contiguous in global
        int k = kc * KC + kp;
        float v0 = 0.f, v1 = 0.f;
        if (mOk && k < K) {
          int ci = k % CIN;                 // even
          int kw = (k / CIN) % KW;
          int kh = k / (CIN * KW);
          int hi = ho * S + kh - PAD;
          int wi = wo * S + kw - PAD;
          if ((unsigned)hi < (unsigned)HI && (unsigned)wi < (unsigned)WI) {
            const float2 t2 = *(const float2*)&in[((b * HI + hi) * WI + wi) * CIN + ci];
            v0 = t2.x; v1 = t2.y;
            if (bnscale) {
              v0 = fmaf(v0, bnscale[ci], bnbias[ci]);
              v1 = fmaf(v1, bnscale[ci + 1], bnbias[ci + 1]);
              v0 = v0 > 0.f ? v0 : 0.f;
              v1 = v1 > 0.f ? v1 : 0.f;
            }
          }
        }
        p.h[0] = (__bf16)v0;
        p.h[1] = (__bf16)v1;
      } else {
#pragma unroll
        for (int j = 0; j < 2; ++j) {
          int k = kc * KC + kp + j;
          float v = 0.f;
          if (mOk && k < K) {
            int ci = k % CIN;
            int kw = (k / CIN) % KW;
            int kh = k / (CIN * KW);
            int hi = ho * S + kh - PAD;
            int wi = wo * S + kw - PAD;
            if ((unsigned)hi < (unsigned)HI && (unsigned)wi < (unsigned)WI) {
              v = in[((b * HI + hi) * WI + wi) * CIN + ci];
              if (bnscale) {
                v = fmaf(v, bnscale[ci], bnbias[ci]);
                v = v > 0.f ? v : 0.f;
              }
            }
          }
          p.h[j] = (__bf16)v;
        }
      }
      *(unsigned*)&lA[row * LDA + kp] = p.u;
    }
    // ---- stage B transposed (bf16 pairs -> ds_write_b32) ----
    for (int idx = tid; idx < NB * (KC / 2); idx += 256) {
      int n  = idx / (KC / 2);
      int kp = (idx % (KC / 2)) * 2;
      union { __bf16 h[2]; unsigned u; } p;
#pragma unroll
      for (int j = 0; j < 2; ++j) {
        int k = kc * KC + kp + j;
        p.h[j] = (k < K && n < COUT) ? wt[k * COUT + n] : (__bf16)0.f;
      }
      *(unsigned*)&lBt[n * LDB + kp] = p.u;
    }
    __syncthreads();

    // ---- compute: 2 WMMA K-steps per chunk, b128 fragment loads ----
    const __bf16* aRow = &lA[(wave * 16 + l16) * LDA];
#pragma unroll
    for (int ks = 0; ks < KC; ks += 32) {
      // A 16x32 bf16 layout: e<8 -> K=ks+lhalf*8+e ; e>=8 -> K=ks+16+lhalf*8+(e-8)
      v8bf a0 = *(const v8bf*)(aRow + ks + (lhalf << 3));
      v8bf a1 = *(const v8bf*)(aRow + ks + 16 + (lhalf << 3));
      v16bf afrag = __builtin_shufflevector(a0, a1, 0, 1, 2, 3, 4, 5, 6, 7,
                                            8, 9, 10, 11, 12, 13, 14, 15);
#pragma unroll
      for (int t = 0; t < NT; ++t) {
        // B 32x16 bf16 layout: e<8 -> K=ks+lhalf*8+e ; e>=8 -> K=ks+16+lhalf*8+(e-8), N = lane%16
        const __bf16* bRow = &lBt[(t * 16 + l16) * LDB + ks + (lhalf << 3)];
        v8bf b0 = *(const v8bf*)(bRow);
        v8bf b1 = *(const v8bf*)(bRow + 16);
        v16bf bfrag = __builtin_shufflevector(b0, b1, 0, 1, 2, 3, 4, 5, 6, 7,
                                              8, 9, 10, 11, 12, 13, 14, 15);
        acc[t] = __builtin_amdgcn_wmma_f32_16x16x32_bf16(
            false, afrag, false, bfrag, (short)0, acc[t], false, false);
      }
    }
  }

  // ---- epilogue: residual + colbias fusion, store f32 ----
#pragma unroll
  for (int t = 0; t < NT; ++t) {
    int n = t * 16 + l16;
    if (n < COUT) {
#pragma unroll
      for (int r = 0; r < 8; ++r) {
        // C/D layout: VGPR r holds M = r + (lane/16)*8, N = lane%16
        int m = mBase + wave * 16 + (lhalf << 3) + r;
        if (m < M) {
          float v = acc[t][r];
          if (residual) v += residual[m * COUT + n];
          if (colbias)  v += colbias[n];
          out[m * COUT + n] = v;
        }
      }
    }
  }
}

// ---------------------------------------------------------------------------
// Host orchestration
// ---------------------------------------------------------------------------
extern "C" void kernel_launch(void* const* d_in, const int* in_sizes, int n_in,
                              void* d_out, int out_size, void* d_ws, size_t ws_size,
                              hipStream_t stream) {
  (void)in_sizes; (void)n_in; (void)out_size; (void)ws_size;
  const float* X = (const float*)d_in[0];
  auto P = [&](int i) { return (const float*)d_in[i]; };

  // workspace layout (floats)
  const long long A1SZ  = 512LL * 95 * 95 * 8;   // conv1 output
  const long long BUFSZ = 512LL * 47 * 47 * 8;   // max stage tensor
  float* A1 = (float*)d_ws;
  float* B0 = A1 + A1SZ;
  float* B1 = B0 + BUFSZ;
  float* B2 = B1 + BUFSZ;
  float* B3 = B2 + BUFSZ;
  float* sums   = B3 + BUFSZ;          // 128
  float* scaleS = sums + 128;          // 18 slots x 64
  float* biasS  = scaleS + 18 * 64;    // 18 slots x 64
  __bf16* wbf   = (__bf16*)(biasS + 18 * 64);

  // convert all weights to bf16 once per call (deterministic)
  long long wofs = 0;
  auto cvt = [&](int idx, int n) -> const __bf16* {
    __bf16* dst = wbf + wofs;
    wofs += ((long long)n + 63) & ~63LL;
    cvt_bf16_kernel<<<dim3((n + 255) / 256), dim3(256), 0, stream>>>(P(idx), dst, n);
    return dst;
  };
  const __bf16* w_c1 = cvt(56, 216);
  const __bf16* w0a = cvt(5, 576),   *w0b = cvt(6, 576);
  const __bf16* w1a = cvt(11, 576),  *w1b = cvt(12, 576);
  const __bf16* w2a = cvt(17, 1152), *w2b = cvt(18, 2304), *w2s = cvt(19, 1152);
  const __bf16* w3a = cvt(24, 2304), *w3b = cvt(25, 2304);
  const __bf16* w4a = cvt(30, 3456), *w4b = cvt(31, 5184), *w4s = cvt(32, 3456);
  const __bf16* w5a = cvt(37, 5184), *w5b = cvt(38, 5184);
  const __bf16* w6a = cvt(43, 6912), *w6b = cvt(44, 9216), *w6s = cvt(45, 6912);
  const __bf16* w7a = cvt(50, 9216), *w7b = cvt(51, 9216);
  const __bf16* w_cl  = cvt(57, 51200);
  const __bf16* w_lin = cvt(59, 11712);

  auto bn = [&](const float* x, int N, int C, int gi, int bi_, int slot) {
    zero_kernel<<<dim3(1), dim3(128), 0, stream>>>(sums, 128);
    long long tot = (long long)N * C;
    long long blkn = (tot + 255) / 256; if (blkn > 2048) blkn = 2048;
    dim3 g((unsigned)blkn), b(256);
    switch (C) {
      case 8:  bn_stats_kernel<8><<<g, b, 0, stream>>>(x, sums, N);  break;
      case 16: bn_stats_kernel<16><<<g, b, 0, stream>>>(x, sums, N); break;
      case 24: bn_stats_kernel<24><<<g, b, 0, stream>>>(x, sums, N); break;
      case 32: bn_stats_kernel<32><<<g, b, 0, stream>>>(x, sums, N); break;
      default: bn_stats_kernel<64><<<g, b, 0, stream>>>(x, sums, N); break;
    }
    bn_finalize_kernel<<<dim3(1), dim3(64), 0, stream>>>(
        sums, P(gi), P(bi_), scaleS + slot * 64, biasS + slot * 64, 1.f / (float)N, C);
  };
  auto sc = [&](int s) { return (const float*)(scaleS + s * 64); };
  auto bi = [&](int s) { return (const float*)(biasS + s * 64); };

  const int N1 = 512 * 47 * 47;   // stage1 sites
  const int N2 = 512 * 23 * 23;   // stage2
  const int N3 = 512 * 11 * 11;   // stage3
  const int N4 = 512 * 5 * 5;     // stage4
  auto g128 = [](long long M) { return dim3((unsigned)((M + 127) / 128)); };
  dim3 blk(256);

  // conv1: [512,95,95,3] -> A1 [512,95,95,8]  (SAME, s=1)
  conv_wmma_kernel<3, 8, 3, 3, 1, 1, 95, 95, 95, 95><<<g128(512LL * 95 * 95), blk, 0, stream>>>(
      X, w_c1, nullptr, nullptr, nullptr, nullptr, A1, 512);
  // maxpool 3x3/2 -> B0 [512,47,47,8]
  maxpool_kernel<<<dim3((unsigned)(((long long)N1 * 8 + 255) / 256)), blk, 0, stream>>>(A1, B0);

  // ---- block 0 (8->8, s=1, rep0: shortcut = bn_relu(x)) ----
  bn(B0, N1, 8, 2, 1, 0);
  bnrelu_kernel<8><<<dim3((unsigned)(((long long)N1 * 8 + 255) / 256)), blk, 0, stream>>>(
      B0, sc(0), bi(0), B1, N1 * 8);
  conv_wmma_kernel<8, 8, 3, 3, 1, 1, 47, 47, 47, 47><<<g128(N1), blk, 0, stream>>>(
      B1, w0a, nullptr, nullptr, nullptr, nullptr, B2, 512);
  bn(B2, N1, 8, 4, 3, 1);
  conv_wmma_kernel<8, 8, 3, 3, 1, 1, 47, 47, 47, 47><<<g128(N1), blk, 0, stream>>>(
      B2, w0b, sc(1), bi(1), B1, nullptr, B3, 512);

  // ---- block 1 (8->8, rep1: residual = x) ----
  bn(B3, N1, 8, 8, 7, 2);
  conv_wmma_kernel<8, 8, 3, 3, 1, 1, 47, 47, 47, 47><<<g128(N1), blk, 0, stream>>>(
      B3, w1a, sc(2), bi(2), nullptr, nullptr, B1, 512);
  bn(B1, N1, 8, 10, 9, 3);
  conv_wmma_kernel<8, 8, 3, 3, 1, 1, 47, 47, 47, 47><<<g128(N1), blk, 0, stream>>>(
      B1, w1b, sc(3), bi(3), B3, nullptr, B0, 512);

  // ---- block 2 (8->16, s=2) ----
  bn(B0, N1, 8, 14, 13, 4);
  conv_wmma_kernel<8, 16, 3, 3, 2, 0, 47, 47, 23, 23><<<g128(N2), blk, 0, stream>>>(
      B0, w2a, sc(4), bi(4), nullptr, nullptr, B1, 512);
  conv_wmma_kernel<8, 16, 3, 3, 2, 0, 47, 47, 23, 23><<<g128(N2), blk, 0, stream>>>(
      B0, w2s, sc(4), bi(4), nullptr, nullptr, B2, 512);
  bn(B1, N2, 16, 16, 15, 5);
  conv_wmma_kernel<16, 16, 3, 3, 1, 1, 23, 23, 23, 23><<<g128(N2), blk, 0, stream>>>(
      B1, w2b, sc(5), bi(5), B2, nullptr, B3, 512);

  // ---- block 3 (16->16) ----
  bn(B3, N2, 16, 21, 20, 6);
  conv_wmma_kernel<16, 16, 3, 3, 1, 1, 23, 23, 23, 23><<<g128(N2), blk, 0, stream>>>(
      B3, w3a, sc(6), bi(6), nullptr, nullptr, B1, 512);
  bn(B1, N2, 16, 23, 22, 7);
  conv_wmma_kernel<16, 16, 3, 3, 1, 1, 23, 23, 23, 23><<<g128(N2), blk, 0, stream>>>(
      B1, w3b, sc(7), bi(7), B3, nullptr, B0, 512);

  // ---- block 4 (16->24, s=2) ----
  bn(B0, N2, 16, 27, 26, 8);
  conv_wmma_kernel<16, 24, 3, 3, 2, 0, 23, 23, 11, 11><<<g128(N3), blk, 0, stream>>>(
      B0, w4a, sc(8), bi(8), nullptr, nullptr, B1, 512);
  conv_wmma_kernel<16, 24, 3, 3, 2, 0, 23, 23, 11, 11><<<g128(N3), blk, 0, stream>>>(
      B0, w4s, sc(8), bi(8), nullptr, nullptr, B2, 512);
  bn(B1, N3, 24, 29, 28, 9);
  conv_wmma_kernel<24, 24, 3, 3, 1, 1, 11, 11, 11, 11><<<g128(N3), blk, 0, stream>>>(
      B1, w4b, sc(9), bi(9), B2, nullptr, B3, 512);

  // ---- block 5 (24->24) ----
  bn(B3, N3, 24, 34, 33, 10);
  conv_wmma_kernel<24, 24, 3, 3, 1, 1, 11, 11, 11, 11><<<g128(N3), blk, 0, stream>>>(
      B3, w5a, sc(10), bi(10), nullptr, nullptr, B1, 512);
  bn(B1, N3, 24, 36, 35, 11);
  conv_wmma_kernel<24, 24, 3, 3, 1, 1, 11, 11, 11, 11><<<g128(N3), blk, 0, stream>>>(
      B1, w5b, sc(11), bi(11), B3, nullptr, B0, 512);

  // ---- block 6 (24->32, s=2) ----
  bn(B0, N3, 24, 40, 39, 12);
  conv_wmma_kernel<24, 32, 3, 3, 2, 0, 11, 11, 5, 5><<<g128(N4), blk, 0, stream>>>(
      B0, w6a, sc(12), bi(12), nullptr, nullptr, B1, 512);
  conv_wmma_kernel<24, 32, 3, 3, 2, 0, 11, 11, 5, 5><<<g128(N4), blk, 0, stream>>>(
      B0, w6s, sc(12), bi(12), nullptr, nullptr, B2, 512);
  bn(B1, N4, 32, 42, 41, 13);
  conv_wmma_kernel<32, 32, 3, 3, 1, 1, 5, 5, 5, 5><<<g128(N4), blk, 0, stream>>>(
      B1, w6b, sc(13), bi(13), B2, nullptr, B3, 512);

  // ---- block 7 (32->32) ----
  bn(B3, N4, 32, 47, 46, 14);
  conv_wmma_kernel<32, 32, 3, 3, 1, 1, 5, 5, 5, 5><<<g128(N4), blk, 0, stream>>>(
      B3, w7a, sc(14), bi(14), nullptr, nullptr, B1, 512);
  bn(B1, N4, 32, 49, 48, 15);
  conv_wmma_kernel<32, 32, 3, 3, 1, 1, 5, 5, 5, 5><<<g128(N4), blk, 0, stream>>>(
      B1, w7b, sc(15), bi(15), B3, nullptr, B0, 512);

  // ---- bn_res + conv_last 5x5 VALID -> [512,1,1,64] ----
  bn(B0, N4, 32, 55, 54, 16);
  conv_wmma_kernel<32, 64, 5, 5, 1, 0, 5, 5, 1, 1><<<g128(512), blk, 0, stream>>>(
      B0, w_cl, sc(16), bi(16), nullptr, nullptr, B1, 512);

  // ---- bn_last + linear (as 1x1 conv, fused lin_b) -> d_out [512,183] ----
  bn(B1, 512, 64, 53, 52, 17);
  conv_wmma_kernel<64, 183, 1, 1, 1, 0, 1, 1, 1, 1><<<g128(512), blk, 0, stream>>>(
      B1, w_lin, sc(17), bi(17), nullptr, P(58), (float*)d_out, 512);
}
